// SpatialTransformerEncoder_15985868276191
// MI455X (gfx1250) — compile-verified, matrix operand order checked by reading, not run on hardware
//
#include <hip/hip_runtime.h>
#include <hip/hip_bf16.h>
#include <math.h>

// ---------------- problem constants ----------------
#define B_    2048
#define C_    3
#define NKP   32            // keypoints per view
#define E_    256           // embed dim
#define H_    8             // heads
#define HD_   32            // head dim
#define L_    3             // transformer blocks
#define MLP_  1024
#define P_    256
#define KTOK  17            // 16 visible keypoints + view token
#define MROWS 32            // padded token rows for 16x16 WMMA tiling
#define SAMP  (B_*C_)
#define SCALE 0.17677669529663689f   // 32^-0.5

typedef __attribute__((ext_vector_type(16))) _Float16 v16h;
typedef __attribute__((ext_vector_type(8)))  float    v8f;

// ---------------- small device helpers ----------------
__device__ __forceinline__ float gelu_exact(float x) {
  return 0.5f * x * (1.0f + erff(x * 0.70710678118654752f));
}
__device__ __forceinline__ float wave_sum(float v) {
#pragma unroll
  for (int m = 16; m >= 1; m >>= 1) v += __shfl_xor(v, m, 32);
  return v;
}
__device__ __forceinline__ float wave_max(float v) {
#pragma unroll
  for (int m = 16; m >= 1; m >>= 1) v = fmaxf(v, __shfl_xor(v, m, 32));
  return v;
}

// Load one 16x32 f16 operand tile in CDNA5 WMMA lane layout.
// Source is row-major with `stride` halves per row (rows = M for A, = N for
// B when the weight is stored transposed as [N][K]).
// Lane layout (ISA 7.12.2, 16-bit A): lanes 0-15 hold K 0-7 (v[0..7]) and
// K 16-23 (v[8..15]); lanes 16-31 hold K 8-15 and K 24-31.
__device__ __forceinline__ v16h ld_tile(const _Float16* p0, int stride) {
  const int lane = threadIdx.x & 31;
  const _Float16* p = p0 + (lane & 15) * stride + ((lane & 16) ? 8 : 0);
  v16h v;
#pragma unroll
  for (int i = 0; i < 8; ++i) { v[i] = p[i]; v[8 + i] = p[16 + i]; }
  return v;
}

// LayerNorm rows 0..16 of a 32x256 f32 LDS buffer -> f16 LDS buffer,
// rows 17..31 forced to zero (clean WMMA padding). One wave per row.
__device__ void ln_rows(const float* X, const float* g, const float* b,
                        _Float16* dst, int wid, int lane) {
  for (int r = wid; r < MROWS; r += 8) {
    if (r < KTOK) {
      float vals[8]; float s = 0.f, q = 0.f;
#pragma unroll
      for (int j = 0; j < 8; ++j) {
        float v = X[r * E_ + lane * 8 + j];
        vals[j] = v; s += v; q += v * v;
      }
      s = wave_sum(s); q = wave_sum(q);
      float mu  = s * (1.f / E_);
      float var = q * (1.f / E_) - mu * mu;
      float inv = rsqrtf(var + 1e-5f);
#pragma unroll
      for (int j = 0; j < 8; ++j) {
        int e = lane * 8 + j;
        dst[r * E_ + e] = (_Float16)((vals[j] - mu) * inv * g[e] + b[e]);
      }
    } else {
#pragma unroll
      for (int j = 0; j < 8; ++j) dst[r * E_ + lane * 8 + j] = (_Float16)0.f;
    }
  }
}

__device__ float block_sum(float v, float* red, int wid, int lane) {
  v = wave_sum(v);
  __syncthreads();
  if (lane == 0) red[wid] = v;
  __syncthreads();
  float s = 0.f;
#pragma unroll
  for (int i = 0; i < 8; ++i) s += red[i];
  return s;
}

// Generic 32xN = A(32xKD) @ Wt(N x KD)^T + bias GEMM over 16x16 tiles.
// Each wave owns column-tile nt and both row tiles (mt=0,1): every weight
// tile is fetched once and feeds two independent WMMAs (good XDL overlap,
// halves global weight traffic).
// CACHEA: keep the whole A operand (16 tiles = 128 VGPRs) in registers.
// MODE 0: store f16 row-major; MODE 1: gelu then f16; MODE 2: += into f32.
template <int MODE, bool CACHEA, int KD>
__device__ void gemm32(const _Float16* __restrict__ A,
                       const _Float16* __restrict__ Wt, const float* bias,
                       int ncols, _Float16* outH, float* outF, int ostride,
                       int wid, int lane) {
  const int colw = lane & 15;
  const int rofs = (lane & 16) ? 8 : 0;
  v16h ac[16];
  if (CACHEA) {
#pragma unroll
    for (int mt = 0; mt < 2; ++mt)
#pragma unroll
      for (int kk = 0; kk < 8; ++kk)
        ac[mt * 8 + kk] = ld_tile(A + mt * 16 * KD + kk * 32, KD);
  }
  for (int nt = wid; nt < (ncols >> 4); nt += 8) {
    const int nb = nt << 4;
    const float bv = bias[nb + colw];
    v8f acc0, acc1;
#pragma unroll
    for (int i = 0; i < 8; ++i) { acc0[i] = bv; acc1[i] = bv; }
    __builtin_prefetch((const void*)(Wt + ((size_t)nb + 128) * KD), 0, 1);
#pragma unroll
    for (int kk = 0; kk < (KD >> 5); ++kk) {
      v16h w  = ld_tile(Wt + (size_t)nb * KD + kk * 32, KD);
      v16h a0 = CACHEA ? ac[kk]     : ld_tile(A + kk * 32, KD);
      v16h a1 = CACHEA ? ac[8 + kk] : ld_tile(A + 16 * KD + kk * 32, KD);
      acc0 = __builtin_amdgcn_wmma_f32_16x16x32_f16(false, a0, false, w,
                                                    (short)0, acc0, false, false);
      acc1 = __builtin_amdgcn_wmma_f32_16x16x32_f16(false, a1, false, w,
                                                    (short)0, acc1, false, false);
    }
    if (MODE == 1) {
#pragma unroll
      for (int i = 0; i < 8; ++i) { acc0[i] = gelu_exact(acc0[i]); acc1[i] = gelu_exact(acc1[i]); }
    }
    if (MODE == 2) {
      float* pD = outF + nb;
#pragma unroll
      for (int i = 0; i < 8; ++i) {
        pD[(i + rofs) * ostride + colw]      += acc0[i];
        pD[(16 + i + rofs) * ostride + colw] += acc1[i];
      }
    } else {
      _Float16* pD = outH + nb;
#pragma unroll
      for (int i = 0; i < 8; ++i) {
        pD[(i + rofs) * ostride + colw]      = (_Float16)acc0[i];
        pD[(16 + i + rofs) * ostride + colw] = (_Float16)acc1[i];
      }
    }
  }
}

// ---------------- parameter block ----------------
struct Params {
  const float* x; const unsigned char* occ;
  const float* kp_w; const float* kp_b;
  const float* vtok; const float* vpos;
  const float* ln1_s; const float* ln1_b;
  const _Float16* qkv_wt; const float* qkv_b;
  const _Float16* ap_wt;  const float* ap_b;
  const float* ln2_s; const float* ln2_b;
  const _Float16* fc1_wt; const float* fc1_b;
  const _Float16* fc2_wt; const float* fc2_b;
  const float* qvec;
  const float* pln1_s; const float* pln1_b;
  const _Float16* kv_wt;  const float* kv_b;
  const _Float16* pap_wt; const float* pap_b;
  const float* pln2_s; const float* pln2_b;
  const _Float16* pfc1_wt; const float* pfc1_b;
  const _Float16* pfc2_wt; const float* pfc2_b;
  const _Float16* pout_wt; const float* pout_b;
  const float* probe;
  const float* last_s; const float* last_b;
  float* out;
};

// ---------------- prep kernels ----------------
// f32 [R][Cd] -> f16 [Cd][R]  (weight transpose + downconvert, runs once/launch)
__global__ void transpose_to_h(const float* __restrict__ src,
                               _Float16* __restrict__ dst, int R, int Cd) {
  int idx = blockIdx.x * blockDim.x + threadIdx.x;
  if (idx < R * Cd) {
    int r = idx / Cd, cc = idx % Cd;
    dst[(size_t)cc * R + r] = (_Float16)src[idx];
  }
}
// q = probe @ pool_q_w + pool_q_b   (sample-invariant: compute once)
__global__ void qvec_kernel(const float* probe, const float* qw,
                            const float* qb, float* qvec) {
  int e = threadIdx.x;
  float s = qb[e];
  for (int k = 0; k < E_; ++k) s += probe[k] * qw[k * E_ + e];
  qvec[e] = s;
}

// ---------------- fused per-sample transformer ----------------
__global__ __launch_bounds__(256) void stenc_kernel(Params p) {
  const int s    = blockIdx.x;
  const int tid  = threadIdx.x;
  const int lane = tid & 31;
  const int wid  = tid >> 5;
  const int c    = s % C_;

  extern __shared__ char smem[];
  float*    Xres    = (float*)(smem);                  // 32x256 f32  (32 KB)
  _Float16* Aln     = (_Float16*)(smem + 32768);       // 32x256 f16  (16 KB)
  _Float16* Qb      = (_Float16*)(smem + 49152);       // 32x256 f16
  _Float16* Kb      = (_Float16*)(smem + 65536);       // 32x256 f16
  _Float16* Vt      = (_Float16*)(smem + 81920);       // 256x32 f16 (V^T)
  _Float16* Hm      = (_Float16*)(smem + 98304);       // 32x1024 f16 (64 KB)
  float*    scoresF = (float*)(smem + 163840);         // 8 x 32x32 f32 (32 KB)
  _Float16* probsH  = (_Float16*)(smem + 196608);      // 8 x 32x32 f16 (16 KB)
  int*      ordI    = (int*)(smem + 212992);           // 32 ints
  float*    red     = (float*)(smem + 213120);         // 8 floats
  // pool-phase scratch reuses scoresF / probsH
  float*    poolP = scoresF;         // 256
  float*    oBuf  = scoresF + 256;   // 256
  float*    yBuf  = scoresF + 512;   // 256
  float*    y2Buf = scoresF + 768;   // 256
  float*    hBuf  = scoresF + 1024;  // 1024
  _Float16* ynH   = probsH;          // 256 halves

  const int colw = lane & 15;
  const int rofs = (lane & 16) ? 8 : 0;

  // ---- embedding + order-preserving stream compaction ----
  if (tid == 0) {
    int j = 0;
    ordI[j++] = 0;                                  // view token, always visible
    for (int n = 0; n < NKP; ++n)
      if (p.occ[(size_t)s * NKP + n]) ordI[j++] = n + 1;
    for (; j < MROWS; ++j) ordI[j] = -1;
  }
  __syncthreads();
  {
    const int e = tid;                              // one embed column / thread
    for (int r = 0; r < MROWS; ++r) {
      float v = 0.f;
      if (r < KTOK) {
        int t = ordI[r];
        if (t == 0) {
          v = p.vtok[c * E_ + e] + p.vpos[c * E_ + e];
        } else if (t > 0) {
          int n = t - 1;
          float x0 = p.x[((size_t)s * NKP + n) * 2 + 0];
          float x1 = p.x[((size_t)s * NKP + n) * 2 + 1];
          float expo = (float)(2 * (e >> 1)) * (1.f / E_);
          float ang  = (float)n / __powf(10000.f, expo);
          float pe   = (e & 1) ? __cosf(ang) : __sinf(ang);
          v = x0 * p.kp_w[e] + x1 * p.kp_w[E_ + e] + p.kp_b[e] + pe;
        }
      }
      Xres[r * E_ + e] = v;
    }
  }
  __syncthreads();

  // ---- pre-norm transformer blocks ----
  for (int l = 0; l < L_; ++l) {
    ln_rows(Xres, p.ln1_s + l * E_, p.ln1_b + l * E_, Aln, wid, lane);
    __syncthreads();

    // QKV: (32x256) @ (256x768); Q,K row-major, V stored transposed (256x32)
    {
      const _Float16* Wt = p.qkv_wt + (size_t)l * 768 * E_;
      const float* bias  = p.qkv_b + l * 768;
      v16h ac[16];
#pragma unroll
      for (int mt = 0; mt < 2; ++mt)
#pragma unroll
        for (int kk = 0; kk < 8; ++kk)
          ac[mt * 8 + kk] = ld_tile(Aln + mt * 16 * E_ + kk * 32, E_);
      for (int nt = wid; nt < 48; nt += 8) {
        const int nb = nt << 4;
        const float bv = bias[nb + colw];
        v8f acc0, acc1;
#pragma unroll
        for (int i = 0; i < 8; ++i) { acc0[i] = bv; acc1[i] = bv; }
#pragma unroll
        for (int kk = 0; kk < 8; ++kk) {
          v16h w = ld_tile(Wt + (size_t)nb * E_ + kk * 32, E_);
          acc0 = __builtin_amdgcn_wmma_f32_16x16x32_f16(false, ac[kk], false, w,
                                                        (short)0, acc0, false, false);
          acc1 = __builtin_amdgcn_wmma_f32_16x16x32_f16(false, ac[8 + kk], false, w,
                                                        (short)0, acc1, false, false);
        }
        if (nb < 256) {
#pragma unroll
          for (int i = 0; i < 8; ++i) {
            Qb[(i + rofs) * E_ + nb + colw]      = (_Float16)acc0[i];
            Qb[(16 + i + rofs) * E_ + nb + colw] = (_Float16)acc1[i];
          }
        } else if (nb < 512) {
#pragma unroll
          for (int i = 0; i < 8; ++i) {
            Kb[(i + rofs) * E_ + (nb - 256) + colw]      = (_Float16)acc0[i];
            Kb[(16 + i + rofs) * E_ + (nb - 256) + colw] = (_Float16)acc1[i];
          }
        } else {
#pragma unroll
          for (int i = 0; i < 8; ++i) {
            Vt[((nb - 512) + colw) * HD_ + i + rofs]      = (_Float16)acc0[i];
            Vt[((nb - 512) + colw) * HD_ + 16 + i + rofs] = (_Float16)acc1[i];
          }
        }
      }
    }
    __syncthreads();

    // ---- attention: wave `wid` owns head h = wid ----
    {
      const int h = wid;
      // S_h = Q_h K_h^T : 2x2 tiles, head dim 32 = one WMMA k-step
#pragma unroll
      for (int mt = 0; mt < 2; ++mt)
#pragma unroll
        for (int nt = 0; nt < 2; ++nt) {
          v16h a = ld_tile(Qb + mt * 16 * E_ + h * HD_, E_);
          v16h b = ld_tile(Kb + nt * 16 * E_ + h * HD_, E_);
          v8f acc;
#pragma unroll
          for (int i = 0; i < 8; ++i) acc[i] = 0.f;
          acc = __builtin_amdgcn_wmma_f32_16x16x32_f16(false, a, false, b,
                                                       (short)0, acc, false, false);
          float* sp = scoresF + h * 1024 + mt * 16 * 32 + nt * 16;
#pragma unroll
          for (int i = 0; i < 8; ++i) sp[(i + rofs) * 32 + colw] = acc[i];
        }
      __syncthreads();
      // softmax over the 17 valid keys; zero padded rows/cols
      {
        const int r = lane;
        _Float16* prow = probsH + h * 1024 + r * 32;
        if (r < KTOK) {
          const float* srow = scoresF + h * 1024 + r * 32;
          float mx = -1e30f;
          for (int t = 0; t < KTOK; ++t) mx = fmaxf(mx, srow[t] * SCALE);
          float ex[KTOK]; float sum = 0.f;
          for (int t = 0; t < KTOK; ++t) {
            ex[t] = __expf(srow[t] * SCALE - mx); sum += ex[t];
          }
          float inv = 1.f / sum;
          for (int t = 0; t < 32; ++t)
            prow[t] = (t < KTOK) ? (_Float16)(ex[t] * inv) : (_Float16)0.f;
        } else {
          for (int t = 0; t < 32; ++t) prow[t] = (_Float16)0.f;
        }
      }
      __syncthreads();
      // O_h = P V_h : probs (16x32) x V^T rows (32x16), one k-step
#pragma unroll
      for (int mt = 0; mt < 2; ++mt)
#pragma unroll
        for (int nt = 0; nt < 2; ++nt) {
          v16h a = ld_tile(probsH + h * 1024 + mt * 16 * 32, 32);
          v16h b = ld_tile(Vt + (h * HD_ + nt * 16) * 32, 32);
          v8f acc;
#pragma unroll
          for (int i = 0; i < 8; ++i) acc[i] = 0.f;
          acc = __builtin_amdgcn_wmma_f32_16x16x32_f16(false, a, false, b,
                                                       (short)0, acc, false, false);
#pragma unroll
          for (int i = 0; i < 8; ++i)
            Aln[(mt * 16 + i + rofs) * E_ + h * HD_ + nt * 16 + colw] =
                (_Float16)acc[i];
        }
    }
    __syncthreads();

    // proj + residual add into Xres
    gemm32<2, true, E_>(Aln, p.ap_wt + (size_t)l * E_ * E_, p.ap_b + l * E_,
                        E_, nullptr, Xres, E_, wid, lane);
    __syncthreads();

    // MLP
    ln_rows(Xres, p.ln2_s + l * E_, p.ln2_b + l * E_, Aln, wid, lane);
    __syncthreads();
    gemm32<1, true, E_>(Aln, p.fc1_wt + (size_t)l * MLP_ * E_, p.fc1_b + l * MLP_,
                        MLP_, Hm, nullptr, MLP_, wid, lane);
    __syncthreads();
    gemm32<2, false, MLP_>(Hm, p.fc2_wt + (size_t)l * E_ * MLP_, p.fc2_b + l * E_,
                           E_, nullptr, Xres, E_, wid, lane);
    __syncthreads();
  }

  // ---- attention pooling ----
  ln_rows(Xres, p.pln1_s, p.pln1_b, Aln, wid, lane);
  __syncthreads();
  // kv = kn @ pool_kv_w : kk row-major into Kb, vv transposed into Vt
  {
    const _Float16* Wt = p.kv_wt;
    v16h ac[16];
#pragma unroll
    for (int mt = 0; mt < 2; ++mt)
#pragma unroll
      for (int kk = 0; kk < 8; ++kk)
        ac[mt * 8 + kk] = ld_tile(Aln + mt * 16 * E_ + kk * 32, E_);
    for (int nt = wid; nt < 32; nt += 8) {
      const int nb = nt << 4;
      const float bv = p.kv_b[nb + colw];
      v8f acc0, acc1;
#pragma unroll
      for (int i = 0; i < 8; ++i) { acc0[i] = bv; acc1[i] = bv; }
#pragma unroll
      for (int kk = 0; kk < 8; ++kk) {
        v16h w = ld_tile(Wt + (size_t)nb * E_ + kk * 32, E_);
        acc0 = __builtin_amdgcn_wmma_f32_16x16x32_f16(false, ac[kk], false, w,
                                                      (short)0, acc0, false, false);
        acc1 = __builtin_amdgcn_wmma_f32_16x16x32_f16(false, ac[8 + kk], false, w,
                                                      (short)0, acc1, false, false);
      }
      if (nb < 256) {
#pragma unroll
        for (int i = 0; i < 8; ++i) {
          Kb[(i + rofs) * E_ + nb + colw]      = (_Float16)acc0[i];
          Kb[(16 + i + rofs) * E_ + nb + colw] = (_Float16)acc1[i];
        }
      } else {
#pragma unroll
        for (int i = 0; i < 8; ++i) {
          Vt[((nb - 256) + colw) * HD_ + i + rofs]      = (_Float16)acc0[i];
          Vt[((nb - 256) + colw) * HD_ + 16 + i + rofs] = (_Float16)acc1[i];
        }
      }
    }
  }
  __syncthreads();
  // probe attends (M=1 -> scalar; q precomputed): wave h, lane = key token
  {
    const int h = wid;
    float sc = -1e30f;
    if (lane < KTOK) {
      float a = 0.f;
      for (int j = 0; j < HD_; ++j)
        a += p.qvec[h * HD_ + j] * (float)Kb[lane * E_ + h * HD_ + j];
      sc = a * SCALE;
    }
    float mx  = wave_max(sc);
    float ex  = (lane < KTOK) ? __expf(sc - mx) : 0.f;
    float sum = wave_sum(ex);
    poolP[h * 32 + lane] = ex / sum;
  }
  __syncthreads();
  {
    const int h = wid, u = lane;
    float o = 0.f;
    for (int t = 0; t < KTOK; ++t)
      o += poolP[h * 32 + t] * (float)Vt[(h * HD_ + u) * HD_ + t];
    oBuf[h * HD_ + u] = o;
  }
  __syncthreads();
  float yv;
  {
    const int e = tid;
    float a = p.probe[e] + p.pap_b[e];
    for (int k = 0; k < E_; ++k)
      a += oBuf[k] * (float)p.pap_wt[(size_t)e * E_ + k];
    yv = a; yBuf[e] = a;
  }
  __syncthreads();
  {
    float s1 = block_sum(yv, red, wid, lane);
    float s2 = block_sum(yv * yv, red, wid, lane);
    float mu = s1 * (1.f / E_), var = s2 * (1.f / E_) - mu * mu;
    float inv = rsqrtf(var + 1e-5f);
    int e = tid;
    ynH[e] = (_Float16)((yv - mu) * inv * p.pln2_s[e] + p.pln2_b[e]);
  }
  __syncthreads();
  for (int jj = 0; jj < 4; ++jj) {
    int j = tid + jj * 256;
    float a = p.pfc1_b[j];
    for (int k = 0; k < E_; ++k)
      a += (float)ynH[k] * (float)p.pfc1_wt[(size_t)j * E_ + k];
    hBuf[j] = gelu_exact(a);
  }
  __syncthreads();
  float y2;
  {
    const int e = tid;
    float a = yBuf[e] + p.pfc2_b[e];
    for (int j = 0; j < MLP_; ++j)
      a += hBuf[j] * (float)p.pfc2_wt[(size_t)e * MLP_ + j];
    y2 = a; y2Buf[e] = a;
  }
  __syncthreads();
  float ov;
  {
    const int e = tid;
    float a = p.pout_b[e];
    for (int k = 0; k < E_; ++k)
      a += y2Buf[k] * (float)p.pout_wt[(size_t)e * E_ + k];
    ov = a;
  }
  // final LayerNorm over P=256 and write f32 output
  {
    float s1 = block_sum(ov, red, wid, lane);
    float s2 = block_sum(ov * ov, red, wid, lane);
    float mu = s1 * (1.f / P_), var = s2 * (1.f / P_) - mu * mu;
    float inv = rsqrtf(var + 1e-5f);
    int e = tid;
    p.out[(size_t)s * P_ + e] = (ov - mu) * inv * p.last_s[e] + p.last_b[e];
  }
}

// ---------------- host side ----------------
extern "C" void kernel_launch(void* const* d_in, const int* in_sizes, int n_in,
                              void* d_out, int out_size, void* d_ws, size_t ws_size,
                              hipStream_t stream) {
  (void)in_sizes; (void)n_in; (void)out_size; (void)ws_size;
  const float* x        = (const float*)d_in[0];
  const unsigned char* occ = (const unsigned char*)d_in[1];   // bool mask
  const float* kp_w     = (const float*)d_in[2];
  const float* kp_b     = (const float*)d_in[3];
  const float* vtok     = (const float*)d_in[4];
  const float* vpos     = (const float*)d_in[5];
  const float* ln1_s    = (const float*)d_in[6];
  const float* ln1_b    = (const float*)d_in[7];
  const float* qkv_w    = (const float*)d_in[8];
  const float* qkv_b    = (const float*)d_in[9];
  const float* ap_w     = (const float*)d_in[10];
  const float* ap_b     = (const float*)d_in[11];
  const float* ln2_s    = (const float*)d_in[12];
  const float* ln2_b    = (const float*)d_in[13];
  const float* fc1_w    = (const float*)d_in[14];
  const float* fc1_b    = (const float*)d_in[15];
  const float* fc2_w    = (const float*)d_in[16];
  const float* fc2_b    = (const float*)d_in[17];
  const float* probe    = (const float*)d_in[18];
  const float* pln1_s   = (const float*)d_in[19];
  const float* pln1_b   = (const float*)d_in[20];
  const float* pq_w     = (const float*)d_in[21];
  const float* pq_b     = (const float*)d_in[22];
  const float* pkv_w    = (const float*)d_in[23];
  const float* pkv_b    = (const float*)d_in[24];
  const float* pap_w    = (const float*)d_in[25];
  const float* pap_b    = (const float*)d_in[26];
  const float* pln2_s   = (const float*)d_in[27];
  const float* pln2_b   = (const float*)d_in[28];
  const float* pfc1_w   = (const float*)d_in[29];
  const float* pfc1_b   = (const float*)d_in[30];
  const float* pfc2_w   = (const float*)d_in[31];
  const float* pfc2_b   = (const float*)d_in[32];
  const float* pout_w   = (const float*)d_in[33];
  const float* pout_b   = (const float*)d_in[34];
  const float* last_s   = (const float*)d_in[35];
  const float* last_b   = (const float*)d_in[36];

  // workspace: transposed f16 weights + precomputed pooled query vector
  char* ws = (char*)d_ws;
  _Float16* qkv_wt  = (_Float16*)(ws + 0);         // 3*768*256
  _Float16* ap_wt   = (_Float16*)(ws + 1179648);   // 3*256*256
  _Float16* fc1_wt  = (_Float16*)(ws + 1572864);   // 3*1024*256
  _Float16* fc2_wt  = (_Float16*)(ws + 3145728);   // 3*256*1024
  _Float16* kv_wt   = (_Float16*)(ws + 4718592);   // 512*256
  _Float16* pap_wt  = (_Float16*)(ws + 4980736);   // 256*256
  _Float16* pfc1_wt = (_Float16*)(ws + 5111808);   // 1024*256
  _Float16* pfc2_wt = (_Float16*)(ws + 5636096);   // 256*1024
  _Float16* pout_wt = (_Float16*)(ws + 6160384);   // 256*256
  float*    qvec    = (float*)(ws + 6291456);      // 256 f32

  auto tp = [&](const float* src, _Float16* dst, int R, int Cd) {
    int n = R * Cd;
    transpose_to_h<<<(n + 255) / 256, 256, 0, stream>>>(src, dst, R, Cd);
  };
  for (int l = 0; l < L_; ++l) {
    tp(qkv_w + (size_t)l * E_ * 768, qkv_wt + (size_t)l * 768 * E_, E_, 768);
    tp(ap_w  + (size_t)l * E_ * E_,  ap_wt  + (size_t)l * E_ * E_,  E_, E_);
    tp(fc1_w + (size_t)l * E_ * MLP_, fc1_wt + (size_t)l * MLP_ * E_, E_, MLP_);
    tp(fc2_w + (size_t)l * MLP_ * E_, fc2_wt + (size_t)l * E_ * MLP_, MLP_, E_);
  }
  tp(pkv_w,  kv_wt,   E_, 512);
  tp(pap_w,  pap_wt,  E_, E_);
  tp(pfc1_w, pfc1_wt, E_, MLP_);
  tp(pfc2_w, pfc2_wt, MLP_, E_);
  tp(pout_w, pout_wt, E_, P_);
  qvec_kernel<<<1, 256, 0, stream>>>(probe, pq_w, pq_b, qvec);

  Params prm;
  prm.x = x; prm.occ = occ; prm.kp_w = kp_w; prm.kp_b = kp_b;
  prm.vtok = vtok; prm.vpos = vpos;
  prm.ln1_s = ln1_s; prm.ln1_b = ln1_b;
  prm.qkv_wt = qkv_wt; prm.qkv_b = qkv_b;
  prm.ap_wt = ap_wt;   prm.ap_b = ap_b;
  prm.ln2_s = ln2_s; prm.ln2_b = ln2_b;
  prm.fc1_wt = fc1_wt; prm.fc1_b = fc1_b;
  prm.fc2_wt = fc2_wt; prm.fc2_b = fc2_b;
  prm.qvec = qvec;
  prm.pln1_s = pln1_s; prm.pln1_b = pln1_b;
  prm.kv_wt = kv_wt;   prm.kv_b = pkv_b;
  prm.pap_wt = pap_wt; prm.pap_b = pap_b;
  prm.pln2_s = pln2_s; prm.pln2_b = pln2_b;
  prm.pfc1_wt = pfc1_wt; prm.pfc1_b = pfc1_b;
  prm.pfc2_wt = pfc2_wt; prm.pfc2_b = pfc2_b;
  prm.pout_wt = pout_wt; prm.pout_b = pout_b;
  prm.probe = probe; prm.last_s = last_s; prm.last_b = last_b;
  prm.out = (float*)d_out;

  // ~208 KB LDS per workgroup — legal on CDNA5 (320 KB/WGP), one sample/WG.
  const size_t smem_bytes = 213248;
  stenc_kernel<<<SAMP, 256, smem_bytes, stream>>>(prm);
}
